// RecurrentActorCritic_16947940950212
// MI455X (gfx1250) — matrix operand order repaired and checked by reading, not compiled, not run on hardware
//
#include <hip/hip_runtime.h>
#include <cstdint>

#define Bn 512
#define Tn 512
#define Sn 64
#define Hn 128
#define Gn 384   /* 3H */
#define An 8
#define BT (Bn*Tn)
#define LN_EPS 1e-5f

#define SP (Sn+4)   /* padded LDS row strides (bank-conflict avoidance) */
#define HP (Hn+4)

typedef __bf16 bf16_t;
typedef bf16_t v16bf __attribute__((ext_vector_type(16)));
typedef bf16_t v8bf  __attribute__((ext_vector_type(8)));
typedef float  v8f   __attribute__((ext_vector_type(8)));

static __device__ __forceinline__ v8f wmma_bf16(v16bf a, v16bf b, v8f c) {
  return __builtin_amdgcn_wmma_f32_16x16x32_bf16(false, a, false, b, (short)0, c, false, false);
}

static __device__ __forceinline__ float sigm(float x) {
  return 1.f / (1.f + __expf(-x));
}
static __device__ __forceinline__ float tanh_fast(float x) {
  return 2.f / (1.f + __expf(-2.f * x)) - 1.f;
}

// Pure-VALU all-lanes sum within each 16-lane DPP row (row_ror rotations).
template <int CTRL>
static __device__ __forceinline__ float dpp_add(float x) {
  int y = __builtin_amdgcn_update_dpp(0, __float_as_int(x), CTRL, 0xf, 0xf, true);
  return x + __int_as_float(y);
}
static __device__ __forceinline__ float rowsum16(float x) {
  x = dpp_add<0x128>(x);  // row_ror:8
  x = dpp_add<0x124>(x);  // row_ror:4
  x = dpp_add<0x122>(x);  // row_ror:2
  x = dpp_add<0x121>(x);  // row_ror:1
  return x;
}

// A-fragment (16x32 bf16, wave32): row m = lane&15,
// elem e -> k = kbase + 16*(lane/16) + e
static __device__ __forceinline__ v16bf load_a_lds(const float* src, int ld, int lane, int kbase) {
  const int m = lane & 15, hi = lane >> 4;
  v16bf a;
#pragma unroll
  for (int e = 0; e < 16; ++e) {
    int k = kbase + (hi << 4) + e;
    a[e] = (bf16_t)src[m * ld + k];
  }
  return a;
}

// B-fragment (32x16 bf16): col n = nbase + (lane&15), elem e -> k = kbase + 16*(lane/16) + e
// Weight row-major (N x K): value = W[n*ld + k]  (i.e. W^T[k][n]).
static __device__ __forceinline__ v16bf load_bT_global(const float* W, int ld, int lane, int nbase, int kbase) {
  const int n  = nbase + (lane & 15);
  const int k0 = kbase + ((lane >> 4) << 4);
  v16bf b;
#pragma unroll
  for (int e = 0; e < 16; ++e) b[e] = (bf16_t)W[n * ld + k0 + e];
  return b;
}

// ---------------------------------------------------------------------------
// Kernel 1: z = relu(LN(x @ Wf^T + bf)); gx = z @ Wih^T + bih  -> bf16 scratch
// 256 threads = 8 waves; weight fragments resident (1 + 3 n-tiles per wave).
// ---------------------------------------------------------------------------
__global__ __launch_bounds__(256)
void feature_gx_kernel(const float* __restrict__ x,      // (B,T,S)
                       const float* __restrict__ Wf,     // (H,S)
                       const float* __restrict__ bf_,    // (H)
                       const float* __restrict__ gf,     // (H)
                       const float* __restrict__ betf,   // (H)
                       const float* __restrict__ Wih,    // (3H,H)
                       const float* __restrict__ bih,    // (3H)
                       bf16_t* __restrict__ gx)          // (B*T, 3H)
{
  __shared__ float  xs[16*SP];
  __shared__ float  zs[16*HP];
  __shared__ bf16_t gxs[16*Gn];

  const int tid  = threadIdx.x;
  const int lane = tid & 31, wave = tid >> 5;

  // GEMM1 (N=128): wave covers n-tile `wave`
  v16bf wfB[2];
#pragma unroll
  for (int ks = 0; ks < 2; ++ks)
    wfB[ks] = load_bT_global(Wf, Sn, lane, wave*16, ks*32);

  // GEMM2 (N=384): wave covers n-tiles {3w..3w+2}
  v16bf wihB[3][4];
#pragma unroll
  for (int nt = 0; nt < 3; ++nt)
#pragma unroll
    for (int ks = 0; ks < 4; ++ks)
      wihB[nt][ks] = load_bT_global(Wih, Hn, lane, (wave*3+nt)*16, ks*32);

  const int NT = BT / 16;
  for (int tile = blockIdx.x; tile < NT; tile += gridDim.x) {
    const unsigned row0 = (unsigned)tile * 16u;

    for (int i = tid; i < 16*Sn; i += 256) {
      const int m = i >> 6, s = i & 63;
      xs[m*SP + s] = x[row0 * Sn + (unsigned)i];
    }
    __syncthreads();

    // GEMM1: (16x64) x (64x128) -> zs
    {
      v16bf aX[2];
#pragma unroll
      for (int ks = 0; ks < 2; ++ks) aX[ks] = load_a_lds(xs, SP, lane, ks*32);
      v8f acc = {};
      acc = wmma_bf16(aX[0], wfB[0], acc);
      acc = wmma_bf16(aX[1], wfB[1], acc);
      const int n     = wave*16 + (lane & 15);
      const int mBase = (lane >> 4) << 3;
      const float bn  = bf_[n];
#pragma unroll
      for (int r = 0; r < 8; ++r) zs[(mBase + r)*HP + n] = acc[r] + bn;
    }
    __syncthreads();

    // LayerNorm + ReLU: row m = tid/16, 8 cols per thread; DPP row reduction
    {
      const int m = tid >> 4, p = tid & 15, j0 = p * 8;
      float v[8]; float s = 0.f, s2 = 0.f;
#pragma unroll
      for (int i = 0; i < 8; ++i) { v[i] = zs[m*HP + j0 + i]; s += v[i]; s2 += v[i]*v[i]; }
      s  = rowsum16(s);
      s2 = rowsum16(s2);
      const float mu   = s * (1.f/Hn);
      const float rstd = rsqrtf(s2 * (1.f/Hn) - mu*mu + LN_EPS);
#pragma unroll
      for (int i = 0; i < 8; ++i) {
        const int j = j0 + i;
        float t_ = (v[i] - mu) * rstd * gf[j] + betf[j];
        zs[m*HP + j] = t_ > 0.f ? t_ : 0.f;
      }
    }
    __syncthreads();

    // GEMM2: (16x128) x (128x384) -> gxs (bf16 staging)
    {
      v16bf aZ[4];
#pragma unroll
      for (int ks = 0; ks < 4; ++ks) aZ[ks] = load_a_lds(zs, HP, lane, ks*32);
#pragma unroll
      for (int nt = 0; nt < 3; ++nt) {
        v8f acc = {};
#pragma unroll
        for (int ks = 0; ks < 4; ++ks) acc = wmma_bf16(aZ[ks], wihB[nt][ks], acc);
        const int n     = (wave*3+nt)*16 + (lane & 15);
        const int mBase = (lane >> 4) << 3;
        const float bn  = bih[n];
#pragma unroll
        for (int r = 0; r < 8; ++r)
          gxs[(mBase + r)*Gn + n] = (bf16_t)(acc[r] + bn);
      }
    }
    __syncthreads();

    // coalesced b128 stream-out of the 16x384 bf16 tile (contiguous in gx)
    {
      v8bf* __restrict__ dst = (v8bf*)(gx + (size_t)row0 * Gn);
      const v8bf* __restrict__ src = (const v8bf*)gxs;
#pragma unroll
      for (int i = 0; i < 3; ++i) dst[tid + i*256] = src[tid + i*256];
    }
    __syncthreads();
  }
}

// ---------------------------------------------------------------------------
// Kernel 2: persistent GRU scan. 32 blocks x 16 batch rows; h double-buffered
// in LDS; 2 barriers per timestep; W_hh^T fragments register-resident;
// g_r / beta_r folded into head weights; gx/done loads issued at iteration
// top so their latency hides behind fragment build + WMMA + barrier.
// ---------------------------------------------------------------------------
__global__ __launch_bounds__(256)
void gru_scan_kernel(const bf16_t* __restrict__ gx,     // (B*T, 3H)
                     const float* __restrict__ hx,      // (1,B,H)
                     const int*   __restrict__ done,    // (B,T)
                     const float* __restrict__ Whh,     // (3H,H)
                     const float* __restrict__ bhh,     // (3H)
                     const float* __restrict__ gr,      // (H)
                     const float* __restrict__ betr,    // (H)
                     const float* __restrict__ Wpol,    // (A,H)
                     const float* __restrict__ bpol,    // (A)
                     const float* __restrict__ Wval,    // (1,H)
                     const float* __restrict__ bval,    // (1)
                     float* __restrict__ outLogits,     // (B,T,A)
                     float* __restrict__ outVals,       // (B,T)
                     float* __restrict__ outH)          // (B,H)
{
  __shared__ float hS[2][16*HP];
  __shared__ float ghS[16*Gn];
  __shared__ float yS[16*HP];
  __shared__ float lmu[16], lrs[16];

  const int tid  = threadIdx.x;
  const int lane = tid & 31, wave = tid >> 5;
  const int b0   = blockIdx.x * 16;

  // resident W_hh^T fragments: wave covers gh n-tiles {3w..3w+2}
  v16bf whhB[3][4];
#pragma unroll
  for (int nt = 0; nt < 3; ++nt)
#pragma unroll
    for (int ks = 0; ks < 4; ++ks)
      whhB[nt][ks] = load_bT_global(Whh, Hn, lane, (wave*3+nt)*16, ks*32);

  // wave 0: head weights with g_r folded in; bias with beta_r folded in.
  v16bf headB[4];
  float headBias = 0.f;
  if (wave == 0) {
    const int n  = lane & 15;
    const int k0 = (lane >> 4) << 4;
#pragma unroll
    for (int ks = 0; ks < 4; ++ks) {
#pragma unroll
      for (int e = 0; e < 16; ++e) {
        const int k = ks*32 + k0 + e;
        const float w = (n < An) ? Wpol[n*Hn + k] : ((n == An) ? Wval[k] : 0.f);
        headB[ks][e] = (bf16_t)(gr[k] * w);
      }
    }
    if (n < An) {
      float s = bpol[n];
      for (int k = 0; k < Hn; ++k) s += betr[k] * Wpol[n*Hn + k];
      headBias = s;
    } else if (n == An) {
      float s = bval[0];
      for (int k = 0; k < Hn; ++k) s += betr[k] * Wval[k];
      headBias = s;
    }
  }

  for (int i = tid; i < 16*Hn; i += 256) {
    const int m = i >> 7, j = i & 127;
    hS[0][m*HP + j] = hx[(unsigned)b0 * Hn + (unsigned)i];
  }
  __syncthreads();

  // per-thread GRU-phase coordinates (loop-invariant)
  const int gm = tid >> 4, gp = tid & 15, gj0 = gp * 8;
  const unsigned btRow = (unsigned)(b0 + gm) * Tn;

  for (int t = 0; t < Tn; ++t) {
    const float* hCur = hS[t & 1];
    float*       hNxt = hS[(t + 1) & 1];

    // ---- early global loads for this step (overlap with WMMA + barrier) ----
    const unsigned bt   = btRow + (unsigned)t;
    const unsigned base = bt * (unsigned)Gn;
    const v8bf xrv = *(const v8bf*)(gx + base + gj0);
    const v8bf xzv = *(const v8bf*)(gx + base + Hn + gj0);
    const v8bf xnv = *(const v8bf*)(gx + base + 2*Hn + gj0);
    const float dmask = 1.f - (float)done[bt];

    // A fragments of current h (each wave builds its own copy; padded stride)
    v16bf aH[4];
#pragma unroll
    for (int ks = 0; ks < 4; ++ks) aH[ks] = load_a_lds(hCur, HP, lane, ks*32);

    // gh = h @ Whh^T + bhh  -> ghS
#pragma unroll
    for (int nt = 0; nt < 3; ++nt) {
      v8f acc = {};
#pragma unroll
      for (int ks = 0; ks < 4; ++ks) acc = wmma_bf16(aH[ks], whhB[nt][ks], acc);
      const int n     = (wave*3+nt)*16 + (lane & 15);
      const int mBase = (lane >> 4) << 3;
      const float bn  = bhh[n];
#pragma unroll
      for (int r = 0; r < 8; ++r) ghS[(mBase + r)*Gn + n] = acc[r] + bn;
    }
    __syncthreads();   // b1: ghS ready (also fences prev-step ghS/yS readers)

    // fused GRU cell + done mask + LN stats via DPP row reduction
    {
      float s = 0.f, s2 = 0.f;
#pragma unroll
      for (int i = 0; i < 8; ++i) {
        const int j = gj0 + i;
        const float hr  = ghS[gm*Gn + j];
        const float hz  = ghS[gm*Gn + Hn + j];
        const float hn_ = ghS[gm*Gn + 2*Hn + j];
        const float hprev = hCur[gm*HP + j];
        const float r  = sigm((float)xrv[i] + hr);
        const float u  = sigm((float)xzv[i] + hz);
        const float nn = tanh_fast((float)xnv[i] + r * hn_);
        const float hnew = (1.f - u) * nn + u * hprev;
        yS[gm*HP + j]   = hnew;
        hNxt[gm*HP + j] = hnew * dmask;
        s += hnew; s2 += hnew * hnew;
      }
      s  = rowsum16(s);
      s2 = rowsum16(s2);
      if (gp == 0) {
        const float mu = s * (1.f/Hn);
        lmu[gm] = mu;
        lrs[gm] = rsqrtf(s2 * (1.f/Hn) - mu*mu + LN_EPS);
      }
    }
    __syncthreads();   // b2: yS / lmu / lrs / hNxt ready

    // wave 0: policy+value heads as one WMMA n-tile; LN applied during A build
    if (wave == 0) {
      const int m  = lane & 15;
      const int hi = lane >> 4;
      const float mu = lmu[m], rstd = lrs[m];
      v16bf aY[4];
#pragma unroll
      for (int ks = 0; ks < 4; ++ks) {
#pragma unroll
        for (int e = 0; e < 16; ++e) {
          const int k = ks*32 + (hi << 4) + e;
          aY[ks][e] = (bf16_t)((yS[m*HP + k] - mu) * rstd);
        }
      }
      v8f acc = {};
#pragma unroll
      for (int ks = 0; ks < 4; ++ks) acc = wmma_bf16(aY[ks], headB[ks], acc);
      const int n     = lane & 15;
      const int mBase = (lane >> 4) << 3;
      if (n <= An) {
#pragma unroll
        for (int r = 0; r < 8; ++r) {
          const unsigned obt = (unsigned)(b0 + mBase + r) * Tn + (unsigned)t;
          const float v = acc[r] + headBias;
          if (n < An) outLogits[obt * An + n] = v;
          else        outVals[obt] = v;
        }
      }
    }
  }

  // final hidden state (Tn even -> lives in hS[0]); last writes fenced by b2
  for (int i = tid; i < 16*Hn; i += 256) {
    const int m = i >> 7, j = i & 127;
    outH[(unsigned)b0 * Hn + (unsigned)i] = hS[Tn & 1][m*HP + j];
  }
}

extern "C" void kernel_launch(void* const* d_in, const int* in_sizes, int n_in,
                              void* d_out, int out_size, void* d_ws, size_t ws_size,
                              hipStream_t stream) {
  const float* x    = (const float*)d_in[0];
  const float* hx   = (const float*)d_in[1];
  const int*   done = (const int*)d_in[2];
  const float* Wf   = (const float*)d_in[3];
  const float* bf_  = (const float*)d_in[4];
  const float* gf   = (const float*)d_in[5];
  const float* betf = (const float*)d_in[6];
  const float* Wih  = (const float*)d_in[7];
  const float* Whh  = (const float*)d_in[8];
  const float* bih  = (const float*)d_in[9];
  const float* bhh  = (const float*)d_in[10];
  const float* gr   = (const float*)d_in[11];
  const float* betr = (const float*)d_in[12];
  const float* Wpol = (const float*)d_in[13];
  const float* bpol = (const float*)d_in[14];
  const float* Wval = (const float*)d_in[15];
  const float* bval = (const float*)d_in[16];

  bf16_t* gx = (bf16_t*)d_ws;  // B*T*3H bf16 = 192 MiB scratch

  float* outLogits = (float*)d_out;                 // (B,T,A)
  float* outVals   = outLogits + (size_t)BT * An;   // (B,T,1)
  float* outH      = outVals + BT;                  // (1,B,H)

  feature_gx_kernel<<<4096, 256, 0, stream>>>(x, Wf, bf_, gf, betf, Wih, bih, gx);
  gru_scan_kernel<<<Bn/16, 256, 0, stream>>>(gx, hx, done, Whh, bhh, gr, betr,
                                             Wpol, bpol, Wval, bval,
                                             outLogits, outVals, outH);
}
